// Self_Attention_4767413699066
// MI455X (gfx1250) — matrix operand order repaired and checked by reading, not compiled, not run on hardware
//
#include <hip/hip_runtime.h>
#include <hip/hip_bf16.h>
#include <math.h>

// ---------------------------------------------------------------------------
// Self-attention x:[4,4096,1024] f32, W*:[1024,128] -> out:[4,4096,128] f32.
// bf16 WMMA (v_wmma_f32_16x16x32_bf16) everywhere, f32 softmax, flash fusion.
// Flash computes S^T = K @ Q^T and O^T = V^T @ P^T so that the softmax is a
// per-lane operation (query = lane column): no LDS in the inner loop.
// ---------------------------------------------------------------------------

#define BATCH   4
#define SEQ     4096
#define DIN     1024
#define DK      128
#define DV      128

typedef __attribute__((ext_vector_type(16))) __bf16        v16bf;
typedef __attribute__((ext_vector_type(8)))  float         v8f;
typedef __attribute__((ext_vector_type(4)))  unsigned int  u32x4;
typedef __attribute__((ext_vector_type(4)))  float         f32x4;

union BF16Frag {            // 16 bf16 = 32 bytes = 8 VGPRs
    v16bf           v;
    u32x4           q[2];
    unsigned int    w[8];
    unsigned short  u[16];
};

__device__ __forceinline__ unsigned short bf16_bits(float f) {
    return __builtin_bit_cast(unsigned short, (__bf16)f);   // native v_cvt path
}
__device__ __forceinline__ unsigned pack_bf16x2(float lo, float hi) {
    return (unsigned)bf16_bits(lo) | ((unsigned)bf16_bits(hi) << 16);
}

// ---------------------------------------------------------------------------
// Prep 1: x (f32, contiguous) -> bf16, 8 elements per thread, b128 in/out.
// ---------------------------------------------------------------------------
__global__ __launch_bounds__(256) void cvt_x_kernel(
    const float* __restrict__ in, unsigned short* __restrict__ outb)
{
    const size_t i = (size_t)blockIdx.x * 256 + threadIdx.x;  // 8 floats each
    const f32x4* p = (const f32x4*)in + i * 2;
    const f32x4 a = p[0], b = p[1];
    u32x4 o;
    o.x = pack_bf16x2(a.x, a.y);
    o.y = pack_bf16x2(a.z, a.w);
    o.z = pack_bf16x2(b.x, b.y);
    o.w = pack_bf16x2(b.z, b.w);
    ((u32x4*)outb)[i] = o;
}

// ---------------------------------------------------------------------------
// Prep 2: W[k][n] (f32, 1024x128) -> Wt[n][k] (bf16, 128x1024).
// ---------------------------------------------------------------------------
__global__ __launch_bounds__(256) void cvt_wt_kernel(
    const float* __restrict__ W, unsigned short* __restrict__ Wt)
{
    const int o = blockIdx.x * 256 + threadIdx.x;   // 131072 elements
    const int n = o >> 10, k = o & 1023;
    Wt[o] = bf16_bits(W[(size_t)k * DK + n]);
}

// ---------------------------------------------------------------------------
// Proj GEMM: out = (xb @ Wt^T + bias) * scale, all-bf16 operands.
// One workgroup = 16 rows; 8 waves cover the 128 output columns.
// Inner loop per wave: 4x global_load_b128 + 1x v_wmma_f32_16x16x32_bf16.
// ---------------------------------------------------------------------------
template <bool TRANS>
__global__ __launch_bounds__(256) void proj_kernel(
    const unsigned short* __restrict__ xb,   // [b][s][DIN] bf16
    const unsigned short* __restrict__ Wt,   // [n][k] bf16 (transposed W)
    const float* __restrict__ bias,
    unsigned short* __restrict__ out, float scale)
{
    const int tile = blockIdx.x;             // 0 .. BATCH*SEQ/16-1
    const int b    = tile >> 8;
    const int r0   = (tile & 255) << 4;
    const int w    = threadIdx.x >> 5;
    const int lane = threadIdx.x & 31;
    const int lh   = lane >> 4;
    const int nlo  = lane & 15;
    const int n0   = w * 16;

    const unsigned short* xrow = xb + ((size_t)b * SEQ + r0 + nlo) * DIN;
    const unsigned short* wrow = Wt + (size_t)(n0 + nlo) * DIN;

    v8f acc = {};
#pragma unroll 4
    for (int kk = 0; kk < DIN; kk += 32) {
        BF16Frag a, bm;
        a.q[0]  = *(const u32x4*)(xrow + kk + lh * 8);        // K 0-7 / 8-15
        a.q[1]  = *(const u32x4*)(xrow + kk + 16 + lh * 8);   // K 16-23 / 24-31
        bm.q[0] = *(const u32x4*)(wrow + kk + lh * 8);        // K 0-7 / 8-15
        bm.q[1] = *(const u32x4*)(wrow + kk + 16 + lh * 8);   // K 16-23 / 24-31
        acc = __builtin_amdgcn_wmma_f32_16x16x32_bf16(
                  false, a.v, false, bm.v, (short)0, acc, false, false);
    }

    const float bia = bias[n0 + nlo];
    if (TRANS) {
        // out[b][col][s]: C-layout rows v=0..7 are contiguous along s.
        u32x4 st;
        st.x = pack_bf16x2((acc[0] + bia) * scale, (acc[1] + bia) * scale);
        st.y = pack_bf16x2((acc[2] + bia) * scale, (acc[3] + bia) * scale);
        st.z = pack_bf16x2((acc[4] + bia) * scale, (acc[5] + bia) * scale);
        st.w = pack_bf16x2((acc[6] + bia) * scale, (acc[7] + bia) * scale);
        unsigned short* dst =
            out + ((size_t)b * DK + n0 + nlo) * SEQ + r0 + 8 * lh;
        *(u32x4*)dst = st;
    } else {
#pragma unroll
        for (int v = 0; v < 8; ++v) {
            const int row = r0 + v + 8 * lh;
            out[((size_t)b * SEQ + row) * DK + n0 + nlo] =
                bf16_bits((acc[v] + bia) * scale);
        }
    }
}

// ---------------------------------------------------------------------------
// Flash attention, transposed formulation. One workgroup per (batch, 16-query
// tile); wave w owns keys {it*256 + w*32 .. +31}, it = 0..15.
//   S^T tiles:  A = K rows (16 keys x 32 dims), B = resident Q^T fragments
//   O^T accum:  A = V^T rows (16 dv x 32 keys), B = P^T (built via shfl_xor)
// Lane = query column; softmax max needs 1 shuffle, l is per-lane, alpha is
// per-lane. Cross-wave (m, l, O^T) merge through LDS at the end.
// ---------------------------------------------------------------------------
__global__ __launch_bounds__(256) void flash_kernel(
    const unsigned short* __restrict__ Q,   // [b][s][dk] bf16, pre-scaled
    const unsigned short* __restrict__ K,   // [b][s][dk] bf16
    const unsigned short* __restrict__ Vt,  // [b][dv][s] bf16 (transposed)
    float* __restrict__ out)                // [b][s][dv] f32
{
    const int tile = blockIdx.x;
    const int b    = tile >> 8;
    const int q0   = (tile & 255) << 4;
    const int tid  = threadIdx.x;
    const int w    = tid >> 5;
    const int lane = tid & 31;
    const int lh   = lane >> 4;
    const int nlo  = lane & 15;

    __shared__ float Obuf[8 * 16 * 128];
    __shared__ float Mbuf[8 * 16];
    __shared__ float Lbuf[8 * 16];

    const unsigned short* Qb = Q  + ((size_t)b * SEQ + q0) * DK;
    const unsigned short* Kb = K  + (size_t)b * SEQ * DK;
    const unsigned short* Vb = Vt + (size_t)b * DV * SEQ;

    // Q^T as 4 resident B-fragments (col = lane's query, table-order dims)
    BF16Frag qf[4];
    {
        const unsigned short* qrow = Qb + (size_t)nlo * DK;
#pragma unroll
        for (int c = 0; c < 4; ++c) {
            const int d0 = c * 32 + lh * 8;
            qf[c].q[0] = *(const u32x4*)(qrow + d0);
            qf[c].q[1] = *(const u32x4*)(qrow + d0 + 16);
        }
    }

    float mcur = -1e30f, lsum = 0.0f;
    v8f oacc[8];
#pragma unroll
    for (int nb = 0; nb < 8; ++nb) oacc[nb] = (v8f){};

    for (int it = 0; it < 16; ++it) {
        const int k0 = it * 256 + w * 32;

        if (it + 1 < 16) {   // warm L2/WGP$ for next strip
            __builtin_prefetch(Kb + (size_t)(k0 + 256 + nlo) * DK, 0, 1);
            __builtin_prefetch(Vb + (size_t)nlo * SEQ + k0 + 256, 0, 1);
        }

        // ---- S^T: two 16-key x 16-query tiles over 4 dim-chunks ----
        v8f s0 = (v8f){}, s1 = (v8f){};
        const unsigned short* krow0 = Kb + (size_t)(k0 + nlo) * DK;
        const unsigned short* krow1 = Kb + (size_t)(k0 + 16 + nlo) * DK;
#pragma unroll
        for (int c = 0; c < 4; ++c) {
            const int d0 = c * 32 + lh * 8;
            BF16Frag kf0, kf1;
            kf0.q[0] = *(const u32x4*)(krow0 + d0);
            kf0.q[1] = *(const u32x4*)(krow0 + d0 + 16);
            kf1.q[0] = *(const u32x4*)(krow1 + d0);
            kf1.q[1] = *(const u32x4*)(krow1 + d0 + 16);
            s0 = __builtin_amdgcn_wmma_f32_16x16x32_bf16(
                     false, kf0.v, false, qf[c].v, (short)0, s0, false, false);
            s1 = __builtin_amdgcn_wmma_f32_16x16x32_bf16(
                     false, kf1.v, false, qf[c].v, (short)0, s1, false, false);
        }
        // lane's 16 values: s0 = keys k0+8*lh+0..7, s1 = keys k0+16+8*lh+0..7

        // ---- per-lane online softmax (query = nlo column) ----
        float mx = fmaxf(s0[0], s1[0]);
#pragma unroll
        for (int v = 1; v < 8; ++v) mx = fmaxf(mx, fmaxf(s0[v], s1[v]));
        mx = fmaxf(mx, __shfl_xor(mx, 16, 32));   // combine lane-halves
        const float mn    = fmaxf(mcur, mx);
        const float alpha = __expf(mcur - mn);
        mcur = mn;

        float p0[8], p1[8], rs = 0.0f;
#pragma unroll
        for (int v = 0; v < 8; ++v) {
            p0[v] = __expf(s0[v] - mn);
            p1[v] = __expf(s1[v] - mn);
            rs += p0[v] + p1[v];
        }
        lsum = lsum * alpha + rs;                 // per-lane half-partial
#pragma unroll
        for (int nb = 0; nb < 8; ++nb) oacc[nb] *= alpha;

        // ---- build P^T B-fragment (keys 0..31 for this lane's query) ----
        // lh=0 holds keys {0..7}(p0) {16..23}(p1); lh=1 holds {8..15}{24..31}.
        // B needs u[0..15] = keys lh*16 + 0..15 -> one exchange with lane^16.
        BF16Frag pB;
#pragma unroll
        for (int j = 0; j < 4; ++j) {
            const unsigned aw = pack_bf16x2(p0[2 * j], p0[2 * j + 1]);
            const unsigned bw = pack_bf16x2(p1[2 * j], p1[2 * j + 1]);
            const unsigned t  = lh ? aw : bw;       // send what partner needs
            const unsigned r  = (unsigned)__shfl_xor((int)t, 16, 32);
            pB.w[j]     = lh ? r : aw;              // keys lh*16 + 0..7
            pB.w[4 + j] = lh ? bw : r;              // keys lh*16 + 8..15
        }

        // ---- O^T += V^T_tile @ P^T over dv = 8 x 16 ----
#pragma unroll
        for (int nb = 0; nb < 8; ++nb) {
            const unsigned short* vrow =
                Vb + (size_t)(nb * 16 + nlo) * SEQ + k0;
            BF16Frag vf;
            vf.q[0] = *(const u32x4*)(vrow + lh * 16);
            vf.q[1] = *(const u32x4*)(vrow + lh * 16 + 8);
            oacc[nb] = __builtin_amdgcn_wmma_f32_16x16x32_bf16(
                           false, vf.v, false, pB.v, (short)0, oacc[nb], false, false);
        }
    }

    // ---- cross-wave log-sum-exp merge through LDS ----
    const float ltot = lsum + __shfl_xor(lsum, 16, 32);
    if (lh == 0) { Mbuf[w * 16 + nlo] = mcur; Lbuf[w * 16 + nlo] = ltot; }

    float* ob = Obuf + w * 16 * 128 + nlo * 128;   // [query][dv]
#pragma unroll
    for (int nb = 0; nb < 8; ++nb) {
        f32x4 lo4 = { oacc[nb][0], oacc[nb][1], oacc[nb][2], oacc[nb][3] };
        f32x4 hi4 = { oacc[nb][4], oacc[nb][5], oacc[nb][6], oacc[nb][7] };
        *(f32x4*)(ob + nb * 16 + 8 * lh)     = lo4;
        *(f32x4*)(ob + nb * 16 + 8 * lh + 4) = hi4;
    }
    __syncthreads();

    float* orow = out + ((size_t)b * SEQ + q0) * DV;
    for (int e = tid; e < 16 * 128; e += 256) {
        const int r = e >> 7, c = e & 127;
        float M = -1e30f;
#pragma unroll
        for (int ww = 0; ww < 8; ++ww) M = fmaxf(M, Mbuf[ww * 16 + r]);
        float L = 0.0f, val = 0.0f;
#pragma unroll
        for (int ww = 0; ww < 8; ++ww) {
            const float a = __expf(Mbuf[ww * 16 + r] - M);
            L   += a * Lbuf[ww * 16 + r];
            val += a * Obuf[ww * 16 * 128 + r * 128 + c];
        }
        orow[(size_t)r * DV + c] = val / L;
    }
}

// ---------------------------------------------------------------------------
extern "C" void kernel_launch(void* const* d_in, const int* in_sizes, int n_in,
                              void* d_out, int out_size, void* d_ws, size_t ws_size,
                              hipStream_t stream) {
    const float* x  = (const float*)d_in[0];
    const float* Wq = (const float*)d_in[1];
    const float* bq = (const float*)d_in[2];
    const float* Wk = (const float*)d_in[3];
    const float* bk = (const float*)d_in[4];
    const float* Wv = (const float*)d_in[5];
    const float* bv = (const float*)d_in[6];
    float* out = (float*)d_out;

    const size_t xe   = (size_t)BATCH * SEQ * DIN;    // 16M
    const size_t qkve = (size_t)BATCH * SEQ * DK;     // 2M
    const size_t we   = (size_t)DIN * DK;             // 128K

    unsigned short* xbf = (unsigned short*)d_ws;
    unsigned short* Qw  = xbf + xe;
    unsigned short* Kw  = Qw + qkve;
    unsigned short* Vw  = Kw + qkve;
    unsigned short* Wtq = Vw + qkve;
    unsigned short* Wtk = Wtq + we;
    unsigned short* Wtv = Wtk + we;

    const float scale = 0.08838834764831845f;         // 1/sqrt(128)
    const dim3 block(256);

    cvt_x_kernel<<<dim3(xe / (256 * 8)), block, 0, stream>>>(x, xbf);
    cvt_wt_kernel<<<dim3(we / 256), block, 0, stream>>>(Wq, Wtq);
    cvt_wt_kernel<<<dim3(we / 256), block, 0, stream>>>(Wk, Wtk);
    cvt_wt_kernel<<<dim3(we / 256), block, 0, stream>>>(Wv, Wtv);

    const dim3 grid(BATCH * SEQ / 16);
    proj_kernel<false><<<grid, block, 0, stream>>>(xbf, Wtq, bq, Qw, scale);
    proj_kernel<false><<<grid, block, 0, stream>>>(xbf, Wtk, bk, Kw, 1.0f);
    proj_kernel<true ><<<grid, block, 0, stream>>>(xbf, Wtv, bv, Vw, 1.0f);
    flash_kernel<<<grid, block, 0, stream>>>(Qw, Kw, Vw, out);
}
